// SFTA_71940702208222
// MI455X (gfx1250) — matrix operand order repaired and checked by reading, not compiled, not run on hardware
//
#include <hip/hip_runtime.h>
#include <hip/hip_bf16.h>

// ---------------------------------------------------------------------------
// Problem constants (from the reference)
// ---------------------------------------------------------------------------
#define BATCH 64
#define CDIM  256
#define HH    48
#define WW    48
#define NPIX  (HH * WW)        // 2304
#define NHEAD 8
#define DHEAD 32               // CDIM / NHEAD
#define MCLS  64               // base_class_num
#define DFF   512              // 2*CDIM

typedef __attribute__((ext_vector_type(16))) __bf16 v16bf;
typedef __attribute__((ext_vector_type(8)))  float  v8f;

// Async global->LDS (CDNA5). Guarded so the file compiles on toolchains that
// don't declare the builtins; fallback is a plain vectorized LDS copy.
#if defined(__AMDGCN__) &&                                                  \
    __has_builtin(__builtin_amdgcn_global_load_async_to_lds_b128) &&        \
    __has_builtin(__builtin_amdgcn_s_wait_asynccnt)
#define SFTA_ASYNC_LDS 1
// The builtin wants v4i (16-byte int vector) pointers in AS1/AS3.
typedef int sfta_v4i __attribute__((vector_size(16)));
typedef __attribute__((address_space(1))) sfta_v4i* sfta_gp;
typedef __attribute__((address_space(3))) sfta_v4i* sfta_lp;
#else
#define SFTA_ASYNC_LDS 0
#endif

// ---------------------------------------------------------------------------
// WMMA helpers (CDNA5 v_wmma_f32_16x16x32_bf16, wave32)
//
// A-matrix (16 rows x 32 K, bf16): lane L<16 -> row M=L holds K=0..7 (v0..3)
//   and K=16..23 (v4..7); lane L>=16 -> row M=L-16 holds K=8..15 / K=24..31.
// B-matrix (32 K x 16 cols, bf16): lane -> col N=L&15, K order same as A.
// C/D (16x16 f32, 8 VGPRs): elem r -> row M = r + 8*(lane>=16), col = lane&15.
// ---------------------------------------------------------------------------
__device__ __forceinline__ v8f wmma_bf16(v16bf a, v16bf b, v8f c) {
  return __builtin_amdgcn_wmma_f32_16x16x32_bf16(
      /*neg_a=*/false, a, /*neg_b=*/false, b,
      /*c_mod=*/(short)0, c, /*reuse_a=*/false, /*reuse_b=*/false);
}

__device__ __forceinline__ v16bf load_a_frag(const __bf16* p, int ld) {
  // p -> element (row0, k0) of a 16x32 row-major tile, leading dim ld
  const int lane = threadIdx.x & 31;
  const int row  = lane & 15;
  const int kh   = (lane >> 4) & 1;
  v16bf a;
#pragma unroll
  for (int i = 0; i < 16; ++i) {
    const int v = i >> 1, e = i & 1;
    const int k = ((v & 4) ? 16 : 0) + kh * 8 + (v & 3) * 2 + e;
    a[i] = p[(size_t)row * ld + k];
  }
  return a;
}

__device__ __forceinline__ v16bf load_b_frag(const __bf16* p, int ld) {
  // p -> element (k0, col0) of a 32x16 row-major tile (K rows), leading dim ld
  const int lane = threadIdx.x & 31;
  v16bf b;
#pragma unroll
  for (int i = 0; i < 16; ++i) b[i] = p[(size_t)((i < 8 ? 0 : 16) + ((lane >> 4) << 3) + (i & 7)) * ld + (lane & 15)];
  return b;
}

__device__ __forceinline__ v16bf load_b_frag_f32(const float* p, size_t ld) {
  const int lane = threadIdx.x & 31;
  v16bf b;
#pragma unroll
  for (int i = 0; i < 16; ++i) b[i] = (__bf16)p[(size_t)((i < 8 ? 0 : 16) + ((lane >> 4) << 3) + (i & 7)) * ld + (lane & 15)];
  return b;
}

// ---------------------------------------------------------------------------
// Kernel 1: weight conversion / transposition to bf16
// ---------------------------------------------------------------------------
__global__ void sfta_prep_weights(const float* __restrict__ wq,
                                  const float* __restrict__ wm,
                                  const float* __restrict__ w1,
                                  const float* __restrict__ w2,
                                  __bf16* __restrict__ wq_bf,
                                  __bf16* __restrict__ wm_bf,
                                  __bf16* __restrict__ w1t_bf,   // [C][DFF]
                                  __bf16* __restrict__ w2t_bf) { // [DFF][C]
  const int i = blockIdx.x * 256 + threadIdx.x;
  if (i < CDIM * CDIM) {
    wq_bf[i] = (__bf16)wq[i];
    wm_bf[i] = (__bf16)wm[i];
  }
  if (i < DFF * CDIM) {
    const int f = i / CDIM, c = i % CDIM;          // w1 is [DFF][C]
    w1t_bf[(size_t)c * DFF + f] = (__bf16)w1[i];
    const int c2 = i / DFF, f2 = i % DFF;          // w2 is [C][DFF]
    w2t_bf[(size_t)f2 * CDIM + c2] = (__bf16)w2[i];
  }
}

// ---------------------------------------------------------------------------
// Kernel 2: K/V projections (batch-invariant): kp[h][d][m], vpT[h][m][d]
// ---------------------------------------------------------------------------
__global__ void sfta_kvproj(const float* __restrict__ bcw,  // [M][C]
                            const float* __restrict__ wk, const float* __restrict__ bk,
                            const float* __restrict__ wv, const float* __restrict__ bv,
                            __bf16* __restrict__ kp,        // [NHEAD][DHEAD][MCLS]
                            __bf16* __restrict__ vpT) {     // [NHEAD][MCLS][DHEAD]
  const int idx = blockIdx.x * 256 + threadIdx.x;
  if (idx >= 2 * CDIM * MCLS) return;
  const int proj = idx >> 14;            // C*M == 16384
  const int r = idx & 16383;
  const int o = r >> 6, m = r & 63;
  const float* wrow = (proj ? wv : wk) + (size_t)o * CDIM;
  const float* crow = bcw + (size_t)m * CDIM;
  float s = (proj ? bv : bk)[o];
#pragma unroll 8
  for (int c = 0; c < CDIM; ++c) s += wrow[c] * crow[c];
  const int d = o >> 3, h = o & 7;       // channel o = d*NHEAD + h
  if (proj == 0) kp[((size_t)h * DHEAD + d) * MCLS + m] = (__bf16)s;
  else           vpT[((size_t)h * MCLS + m) * DHEAD + d] = (__bf16)s;
}

// ---------------------------------------------------------------------------
// Kernel 3/5: 256x256 x [256 x N] projection GEMM via WMMA.
// MODE 0: in = feature f32 [B][C][N]  -> qattn bf16 [B][H][N][DH]
// MODE 1: in = x bf16 [B][C][N]       -> merged f32 [B][N][C]
// grid (N/64, CDIM/128, B), block 256 (8 waves, each 16 o-rows x 64 n-cols)
// ---------------------------------------------------------------------------
template <int MODE>
__global__ void sfta_proj_gemm(const void* __restrict__ xin,
                               const __bf16* __restrict__ Wbf,  // [C][C]
                               const float* __restrict__ bias,
                               void* __restrict__ outp) {
  const int wave = threadIdx.x >> 5, lane = threadIdx.x & 31;
  const int b = blockIdx.z;
  const int o0 = blockIdx.y * 128 + wave * 16;
  const int nb = blockIdx.x * 64;

  v8f acc[4] = {};
#pragma unroll
  for (int kc = 0; kc < 8; ++kc) {
    const v16bf a = load_a_frag(Wbf + (size_t)o0 * CDIM + kc * 32, CDIM);
#pragma unroll
    for (int nt = 0; nt < 4; ++nt) {
      v16bf bf;
      if (MODE == 0) {
        const float* src = (const float*)xin +
            ((size_t)b * CDIM + kc * 32) * NPIX + nb + nt * 16;
        if (kc < 7) __builtin_prefetch(src + 32 * NPIX, 0, 0);
        bf = load_b_frag_f32(src, NPIX);
      } else {
        const __bf16* src = (const __bf16*)xin +
            ((size_t)b * CDIM + kc * 32) * NPIX + nb + nt * 16;
        if (kc < 7) __builtin_prefetch(src + 32 * NPIX, 0, 0);
        bf = load_b_frag(src, NPIX);
      }
      acc[nt] = wmma_bf16(a, bf, acc[nt]);
    }
  }
#pragma unroll
  for (int nt = 0; nt < 4; ++nt) {
#pragma unroll
    for (int r = 0; r < 8; ++r) {
      const int o = o0 + r + ((lane >> 4) << 3);
      const int n = nb + nt * 16 + (lane & 15);
      const float v = acc[nt][r] + bias[o];
      if (MODE == 0) {
        // channel o = d*8 + h  ->  q[b][h][n][d]
        ((__bf16*)outp)[(((size_t)b * NHEAD + (o & 7)) * NPIX + n) * DHEAD + (o >> 3)] =
            (__bf16)v;
      } else {
        ((float*)outp)[((size_t)b * NPIX + n) * CDIM + o] = v;
      }
    }
  }
}

// ---------------------------------------------------------------------------
// Kernel 4: fused attention (scores -> softmax -> PV) per (b, h, 16-row tile)
// grid (N/128, NHEAD, B), block 256 = 8 waves, one 16-row n-tile per wave.
// kp/vpT head tiles (4KB each) staged to LDS with async global->LDS loads.
// ---------------------------------------------------------------------------
__global__ void sfta_attention(const __bf16* __restrict__ qattn, // [B][H][N][DH]
                               const __bf16* __restrict__ kp,    // [H][DH][M]
                               const __bf16* __restrict__ vpT,   // [H][M][DH]
                               __bf16* __restrict__ xbf) {       // [B][C][N]
  const int wave = threadIdx.x >> 5, lane = threadIdx.x & 31;
  const int b = blockIdx.z, h = blockIdx.y;
  const int n0 = blockIdx.x * 128 + wave * 16;

  __shared__ float  ssc[8][16][MCLS];
  __shared__ __bf16 prb[8][16][MCLS];
  __shared__ __bf16 kps[DHEAD * MCLS];   // [d][m], ld = MCLS
  __shared__ __bf16 vps[MCLS * DHEAD];   // [m][d], ld = DHEAD

  // ---- stage K/V head tiles into LDS (async DMA path on CDNA5) ----
  {
    const __bf16* gk = kp  + (size_t)h * DHEAD * MCLS;
    const __bf16* gv = vpT + (size_t)h * MCLS * DHEAD;
#if SFTA_ASYNC_LDS
    const char* gks = (const char*)gk + (size_t)threadIdx.x * 16;
    const char* gvs = (const char*)gv + (size_t)threadIdx.x * 16;
    char* lks = (char*)kps + (size_t)threadIdx.x * 16;
    char* lvs = (char*)vps + (size_t)threadIdx.x * 16;
    __builtin_amdgcn_global_load_async_to_lds_b128(
        (sfta_gp)gks, (sfta_lp)lks, 0, 0);
    __builtin_amdgcn_global_load_async_to_lds_b128(
        (sfta_gp)gvs, (sfta_lp)lvs, 0, 0);
    __builtin_amdgcn_s_wait_asynccnt(0);
#else
    ((int4*)kps)[threadIdx.x] = ((const int4*)gk)[threadIdx.x];
    ((int4*)vps)[threadIdx.x] = ((const int4*)gv)[threadIdx.x];
#endif
  }
  __syncthreads();

  // ---- scores = (Q^T K) / sqrt(DH) ----
  const v16bf aq = load_a_frag(
      qattn + (((size_t)b * NHEAD + h) * NPIX + n0) * DHEAD, DHEAD);
#pragma unroll
  for (int mt = 0; mt < 4; ++mt) {
    const v16bf bk = load_b_frag(kps + mt * 16, MCLS);
    v8f s = {};
    s = wmma_bf16(aq, bk, s);
#pragma unroll
    for (int r = 0; r < 8; ++r) {
      const int row = r + ((lane >> 4) << 3);
      ssc[wave][row][mt * 16 + (lane & 15)] = s[r] * 0.17677669529663687f;
    }
  }
  __syncthreads();

  // ---- softmax over M=64 (one lane per row) ----
  if (lane < 16) {
    float mx = -3.0e38f;
#pragma unroll
    for (int m = 0; m < MCLS; ++m) mx = fmaxf(mx, ssc[wave][lane][m]);
    float sum = 0.f;
#pragma unroll
    for (int m = 0; m < MCLS; ++m) {
      const float e = __expf(ssc[wave][lane][m] - mx);
      ssc[wave][lane][m] = e;
      sum += e;
    }
    const float inv = 1.f / sum;
#pragma unroll
    for (int m = 0; m < MCLS; ++m)
      prb[wave][lane][m] = (__bf16)(ssc[wave][lane][m] * inv);
  }
  __syncthreads();

  // ---- x = prob @ V^T : (16n x 64m) x (64m x 32d) ----
  const v16bf a0 = load_a_frag(&prb[wave][0][0], MCLS);
  const v16bf a1 = load_a_frag(&prb[wave][0][32], MCLS);
#pragma unroll
  for (int dt = 0; dt < 2; ++dt) {
    v8f acc = {};
    const v16bf b0 = load_b_frag(vps + 0 * DHEAD + dt * 16, DHEAD);
    const v16bf b1 = load_b_frag(vps + 32 * DHEAD + dt * 16, DHEAD);
    acc = wmma_bf16(a0, b0, acc);
    acc = wmma_bf16(a1, b1, acc);
#pragma unroll
    for (int r = 0; r < 8; ++r) {
      const int n = n0 + r + ((lane >> 4) << 3);
      const int d = dt * 16 + (lane & 15);
      const int c = d * NHEAD + h;
      xbf[((size_t)b * CDIM + c) * NPIX + n] = (__bf16)acc[r];
    }
  }
}

// ---------------------------------------------------------------------------
// Kernel 6: SFTA cosine re-weighting. One wave per (b,n) row.
// grid B*N/8, block 256.
// ---------------------------------------------------------------------------
__global__ void sfta_cosine(const float* __restrict__ feat,      // [B][C][N]
                            const float* __restrict__ merged_nc, // [B][N][C]
                            float* __restrict__ out_f,           // [B][N][C]
                            __bf16* __restrict__ out_b) {        // [B][N][C]
  const int wave = threadIdx.x >> 5, lane = threadIdx.x & 31;
  const size_t rowid = (size_t)blockIdx.x * 8 + wave;   // 0 .. B*N-1
  const int b = (int)(rowid / NPIX), n = (int)(rowid % NPIX);
  const float* mrow = merged_nc + rowid * CDIM;

  float fv[8];
  float saf = 0.f, saa = 0.f, sff = 0.f;
#pragma unroll
  for (int j = 0; j < 8; ++j) {
    const int c = lane + 32 * j;
    const float f = feat[((size_t)b * CDIM + c) * NPIX + n];
    const float a = mrow[c];
    fv[j] = f;
    saf += a * f; saa += a * a; sff += f * f;
  }
#pragma unroll
  for (int off = 16; off > 0; off >>= 1) {
    saf += __shfl_xor(saf, off, 32);
    saa += __shfl_xor(saa, off, 32);
    sff += __shfl_xor(sff, off, 32);
  }
  const float cosv = saf / ((sqrtf(saa) + 1e-5f) * (sqrtf(sff) + 1e-5f));
  const float scale = cosv + 1.f;
#pragma unroll
  for (int j = 0; j < 8; ++j) {
    const int c = lane + 32 * j;
    const float o = fv[j] * scale;
    out_f[rowid * CDIM + c] = o;
    out_b[rowid * CDIM + c] = (__bf16)o;
  }
}

// ---------------------------------------------------------------------------
// Kernel 7: fused FFN (relu MLP) + residual + LayerNorm + transposed store.
// grid (N/16, B), block 128 = 4 waves. h1 tile (16x512) lives in LDS.
// ---------------------------------------------------------------------------
__global__ void sfta_ffn_ln(const __bf16* __restrict__ out_bf, // [B][N][C]
                            const float* __restrict__ out_f,   // [B][N][C]
                            const __bf16* __restrict__ w1t,    // [C][DFF]
                            const float* __restrict__ b1,
                            const __bf16* __restrict__ w2t,    // [DFF][C]
                            const float* __restrict__ b2v,
                            const float* __restrict__ gamma,
                            const float* __restrict__ beta,
                            float* __restrict__ dout) {        // [B][C][N]
  const int wave = threadIdx.x >> 5, lane = threadIdx.x & 31;
  const int b = blockIdx.y, n0 = blockIdx.x * 16;

  __shared__ __bf16 h1bf[16][DFF];
  __shared__ float  srcf[16][CDIM];
  __shared__ float  red[16][8];
  __shared__ float  rmu[16], rvar[16];

  // ---- stage 1: h1 = relu(out @ w1^T + b1), 16 x 512 ----
  v16bf afr[8];
  const __bf16* abase = out_bf + ((size_t)b * NPIX + n0) * CDIM;
#pragma unroll
  for (int kc = 0; kc < 8; ++kc) afr[kc] = load_a_frag(abase + kc * 32, CDIM);

#pragma unroll
  for (int i = 0; i < 8; ++i) {
    const int ft = wave * 8 + i;       // 32 f-tiles / 4 waves
    v8f acc = {};
#pragma unroll
    for (int kc = 0; kc < 8; ++kc) {
      const __bf16* src = w1t + (size_t)(kc * 32) * DFF + ft * 16;
      __builtin_prefetch(src + 32 * DFF, 0, 0);
      acc = wmma_bf16(afr[kc], load_b_frag(src, DFF), acc);
    }
#pragma unroll
    for (int r = 0; r < 8; ++r) {
      const int row = r + ((lane >> 4) << 3);
      const int col = ft * 16 + (lane & 15);
      float v = acc[r] + b1[col];
      v = v > 0.f ? v : 0.f;
      h1bf[row][col] = (__bf16)v;
    }
  }
  __syncthreads();

  // ---- stage 2: src = out + h1 @ w2^T + b2, 16 x 256 ----
#pragma unroll
  for (int i = 0; i < 4; ++i) {
    const int ct = wave * 4 + i;       // 16 c-tiles / 4 waves
    v8f acc = {};
#pragma unroll
    for (int kf = 0; kf < 16; ++kf) {
      const v16bf a2 = load_a_frag(&h1bf[0][kf * 32], DFF);
      const v16bf bb = load_b_frag(w2t + (size_t)(kf * 32) * CDIM + ct * 16, CDIM);
      acc = wmma_bf16(a2, bb, acc);
    }
#pragma unroll
    for (int r = 0; r < 8; ++r) {
      const int row = r + ((lane >> 4) << 3);
      const int c = ct * 16 + (lane & 15);
      const int n = n0 + row;
      srcf[row][c] = acc[r] + b2v[c] + out_f[((size_t)b * NPIX + n) * CDIM + c];
    }
  }
  __syncthreads();

  // ---- stage 3: LayerNorm per row (256 ch), write transposed to [B][C][N] ----
  const int r = threadIdx.x >> 3;      // 0..15 rows
  const int seg = threadIdx.x & 7;     // 8 segments x 32 channels
  float s = 0.f;
#pragma unroll
  for (int j = 0; j < 32; ++j) s += srcf[r][seg * 32 + j];
  red[r][seg] = s;
  __syncthreads();
  if (seg == 0) {
    float m = 0.f;
#pragma unroll
    for (int k = 0; k < 8; ++k) m += red[r][k];
    rmu[r] = m * (1.f / CDIM);
  }
  __syncthreads();
  const float mu = rmu[r];
  s = 0.f;
#pragma unroll
  for (int j = 0; j < 32; ++j) {
    const float d = srcf[r][seg * 32 + j] - mu;
    s += d * d;
  }
  red[r][seg] = s;
  __syncthreads();
  if (seg == 0) {
    float v = 0.f;
#pragma unroll
    for (int k = 0; k < 8; ++k) v += red[r][k];
    rvar[r] = v * (1.f / CDIM);
  }
  __syncthreads();
  const float inv = rsqrtf(rvar[r] + 1e-5f);
  const int n = n0 + r;
#pragma unroll
  for (int j = 0; j < 32; ++j) {
    const int c = seg * 32 + j;
    dout[((size_t)b * CDIM + c) * NPIX + n] =
        (srcf[r][c] - mu) * inv * gamma[c] + beta[c];
  }
}

// ---------------------------------------------------------------------------
// Host launcher
// ---------------------------------------------------------------------------
extern "C" void kernel_launch(void* const* d_in, const int* in_sizes, int n_in,
                              void* d_out, int out_size, void* d_ws, size_t ws_size,
                              hipStream_t stream) {
  (void)in_sizes; (void)n_in; (void)out_size; (void)ws_size;

  const float* feat  = (const float*)d_in[0];
  const float* bcw   = (const float*)d_in[1];
  const float* wq    = (const float*)d_in[2];
  const float* bq    = (const float*)d_in[3];
  const float* wk    = (const float*)d_in[4];
  const float* bk    = (const float*)d_in[5];
  const float* wv    = (const float*)d_in[6];
  const float* bv    = (const float*)d_in[7];
  const float* wm    = (const float*)d_in[8];
  const float* bm    = (const float*)d_in[9];
  const float* w1    = (const float*)d_in[10];
  const float* b1    = (const float*)d_in[11];
  const float* w2    = (const float*)d_in[12];
  const float* b2    = (const float*)d_in[13];
  const float* gamma = (const float*)d_in[14];
  const float* beta  = (const float*)d_in[15];

  char* p = (char*)d_ws;
  auto carve = [&](size_t bytes) -> void* {
    void* r = (void*)p;
    p += (bytes + 255) & ~(size_t)255;
    return r;
  };

  __bf16* wq_bf  = (__bf16*)carve((size_t)CDIM * CDIM * 2);
  __bf16* wm_bf  = (__bf16*)carve((size_t)CDIM * CDIM * 2);
  __bf16* w1t_bf = (__bf16*)carve((size_t)CDIM * DFF * 2);
  __bf16* w2t_bf = (__bf16*)carve((size_t)DFF * CDIM * 2);
  __bf16* kp     = (__bf16*)carve((size_t)NHEAD * DHEAD * MCLS * 2);
  __bf16* vpT    = (__bf16*)carve((size_t)NHEAD * MCLS * DHEAD * 2);
  __bf16* qattn  = (__bf16*)carve((size_t)BATCH * NHEAD * NPIX * DHEAD * 2);
  __bf16* xbf    = (__bf16*)carve((size_t)BATCH * CDIM * NPIX * 2);
  float*  merged = (float*)carve((size_t)BATCH * NPIX * CDIM * 4);
  float*  out_f  = (float*)carve((size_t)BATCH * NPIX * CDIM * 4);
  __bf16* out_b  = (__bf16*)carve((size_t)BATCH * NPIX * CDIM * 2);

  // 1) weight conversion / transpose
  sfta_prep_weights<<<(DFF * CDIM + 255) / 256, 256, 0, stream>>>(
      wq, wm, w1, w2, wq_bf, wm_bf, w1t_bf, w2t_bf);

  // 2) batch-invariant K/V projections
  sfta_kvproj<<<(2 * CDIM * MCLS + 255) / 256, 256, 0, stream>>>(
      bcw, wk, bk, wv, bv, kp, vpT);

  // 3) Q projection GEMM -> head-major bf16
  sfta_proj_gemm<0><<<dim3(NPIX / 64, CDIM / 128, BATCH), 256, 0, stream>>>(
      (const void*)feat, wq_bf, bq, (void*)qattn);

  // 4) fused attention (async LDS staging of K/V tiles)
  sfta_attention<<<dim3(NPIX / 128, NHEAD, BATCH), 256, 0, stream>>>(
      qattn, kp, vpT, xbf);

  // 5) merge projection GEMM -> merged [B][N][C] f32
  sfta_proj_gemm<1><<<dim3(NPIX / 64, CDIM / 128, BATCH), 256, 0, stream>>>(
      (const void*)xbf, wm_bf, bm, (void*)merged);

  // 6) cosine re-weighting
  sfta_cosine<<<(BATCH * NPIX) / 8, 256, 0, stream>>>(feat, merged, out_f, out_b);

  // 7) fused FFN + residual + LayerNorm -> d_out [B][C][H][W]
  sfta_ffn_ln<<<dim3(NPIX / 16, BATCH), 128, 0, stream>>>(
      out_b, out_f, w1t_bf, b1, w2t_bf, b2, gamma, beta, (float*)d_out);
}
